// GNN_90434831384895
// MI455X (gfx1250) — matrix operand order repaired and checked by reading, not compiled, not run on hardware
//
#include <hip/hip_runtime.h>

// ---------------------------------------------------------------------------
// GGNN forward for MI455X (gfx1250, wave32, WMMA).
// All GEMMs run on v_wmma_f32_16x16x32_bf16 with f32 accumulation and fused
// bias / sigmoid / tanh / GRU-combine epilogues. Each wave computes a 32x32
// output tile (4 WMMAs per K-step, operand fragments reused 2x). Segment
// means use global_atomic_add_f32. Master hidden/edge state kept in f32;
// bf16 mirrors feed the WMMA units. Scratch arena in d_ws (~490 MB).
// ---------------------------------------------------------------------------

typedef __bf16 bf16_t;
typedef __attribute__((ext_vector_type(16))) __bf16 v16bf;
typedef __attribute__((ext_vector_type(8)))  __bf16 v8bf;
typedef __attribute__((ext_vector_type(8)))  float  v8f;

#define HH    512
#define NOBJ  2048
#define NREL  32768
#define OBJC  151
#define RELC  51

// ------------------------------ tiny helpers -------------------------------

__global__ void k_cvt_bf16(const float* __restrict__ s, bf16_t* __restrict__ d, int n) {
    int i = blockIdx.x * blockDim.x + threadIdx.x;
    if (i < n) d[i] = (bf16_t)s[i];
}

__global__ void k_init_state(const float* __restrict__ x, float* __restrict__ hf,
                             bf16_t* __restrict__ hb, bf16_t* __restrict__ ib, int n) {
    int i = blockIdx.x * blockDim.x + threadIdx.x;
    if (i < n) { float v = x[i]; hf[i] = v; hb[i] = (bf16_t)v; ib[i] = (bf16_t)v; }
}

__global__ void k_zero_f32(float* __restrict__ p, int n) {
    for (int i = blockIdx.x * blockDim.x + threadIdx.x; i < n; i += gridDim.x * blockDim.x)
        p[i] = 0.0f;
}

// dst[r, 0:512] = h[rel[r,1]-onb] ; dst[r, 512:1024] = h[rel[r,2]-onb]
__global__ void k_gather_pair(const bf16_t* __restrict__ h, const int* __restrict__ rel,
                              const int* __restrict__ onb, bf16_t* __restrict__ dst, int ld) {
    size_t r = blockIdx.x;
    int s = rel[r * 3 + 1] - onb[0];
    int t = rel[r * 3 + 2] - onb[0];
    for (int c = threadIdx.x; c < HH; c += blockDim.x) {
        dst[r * (size_t)ld + c]       = h[(size_t)s * HH + c];
        dst[r * (size_t)ld + HH + c]  = h[(size_t)t * HH + c];
    }
}

__global__ void k_fill_rows(bf16_t* __restrict__ dst, int ld, const bf16_t* __restrict__ vec) {
    size_t r = blockIdx.x;
    for (int c = threadIdx.x; c < HH; c += blockDim.x) dst[r * (size_t)ld + c] = vec[c];
}

__global__ void k_copy_rows(bf16_t* __restrict__ dst, int ldd,
                            const bf16_t* __restrict__ src, int lds_) {
    size_t r = blockIdx.x;
    for (int c = threadIdx.x; c < HH; c += blockDim.x)
        dst[r * (size_t)ldd + c] = src[r * (size_t)lds_ + c];
}

// column mean over M rows of a [M,512] f32 matrix. grid=16 blocks x 256 thr.
__global__ void k_colmean(const float* __restrict__ X, int M,
                          float* __restrict__ mean, bf16_t* __restrict__ mean_bf) {
    int col = blockIdx.x * 32 + (threadIdx.x & 31);
    int rl  = threadIdx.x >> 5;            // 0..7
    float s = 0.f;
    for (int r = rl; r < M; r += 8) s += X[(size_t)r * HH + col];
    __shared__ float red[256];
    red[threadIdx.x] = s;
    __syncthreads();
    if (rl == 0) {
        float t = 0.f;
        for (int i = 0; i < 8; ++i) t += red[i * 32 + (threadIdx.x & 31)];
        float m = t / (float)M;
        mean[col] = m;
        if (mean_bf) mean_bf[col] = (bf16_t)m;
    }
}

// segment scatter-add of edge rows into src/tgt accumulators + counts
__global__ void k_scatter(const float* __restrict__ edge, const int* __restrict__ rel,
                          const int* __restrict__ onb,
                          float* __restrict__ aggS, float* __restrict__ aggT,
                          float* __restrict__ cS, float* __restrict__ cT) {
    size_t r = blockIdx.x;
    int s = rel[r * 3 + 1] - onb[0];
    int t = rel[r * 3 + 2] - onb[0];
    for (int c = threadIdx.x; c < HH; c += blockDim.x) {
        float v = edge[r * (size_t)HH + c];
        atomicAdd(&aggS[(size_t)s * HH + c], v);
        atomicAdd(&aggT[(size_t)t * HH + c], v);
    }
    if (threadIdx.x == 0) { atomicAdd(&cS[s], 1.f); atomicAdd(&cT[t], 1.f); }
}

__global__ void k_segnorm(const float* __restrict__ agg, const float* __restrict__ cnt,
                          bf16_t* __restrict__ out, int total) {
    int i = blockIdx.x * blockDim.x + threadIdx.x;
    if (i < total) {
        float c = cnt[i >> 9];
        out[i] = (bf16_t)(c > 0.f ? agg[i] / c : 0.f);
    }
}

// matvec y[512] = W[512,K1+K2] @ [x1|x2] + b   (initial global_proj)
__global__ void k_matvec2(const float* __restrict__ W, const float* __restrict__ b,
                          const float* __restrict__ x1, int K1,
                          const float* __restrict__ x2, int K2,
                          float* __restrict__ y, bf16_t* __restrict__ ybf) {
    int t = blockIdx.x * blockDim.x + threadIdx.x;
    if (t >= HH) return;
    const float* row = W + (size_t)t * (K1 + K2);
    float s = b[t];
    for (int k = 0; k < K1; ++k) s += row[k] * x1[k];
    for (int k = 0; k < K2; ++k) s += row[K1 + k] * x2[k];
    y[t] = s;
    if (ybf) ybf[t] = (bf16_t)s;
}

// global-feature GRU (tiny): one block of 512 threads, f32 weights direct.
__global__ void k_global_gru(const float* __restrict__ emean, const float* __restrict__ hmean,
                             const float* __restrict__ w6, const float* __restrict__ b6,
                             const float* __restrict__ u6, const float* __restrict__ bu6,
                             const float* __restrict__ w7, const float* __restrict__ b7,
                             const float* __restrict__ w8, const float* __restrict__ b8,
                             const float* __restrict__ u8, const float* __restrict__ bu8,
                             float* __restrict__ gf, bf16_t* __restrict__ gfb) {
    __shared__ float ev[2 * HH];
    __shared__ float g[HH];
    __shared__ float zu[HH];
    __shared__ float rg[HH];
    int t = threadIdx.x;
    ev[t] = emean[t]; ev[HH + t] = hmean[t]; g[t] = gf[t];
    __syncthreads();
    float s6w = b6[t], s7w = b7[t], s6u = bu6[t];
    for (int k = 0; k < 2 * HH; ++k) { float e = ev[k]; s6w += w6[(size_t)t * 2 * HH + k] * e; s7w += w7[(size_t)t * 2 * HH + k] * e; }
    for (int k = 0; k < HH; ++k) s6u += u6[(size_t)t * HH + k] * g[k];
    float z = 1.f / (1.f + __expf(-(s6w + s6u)));
    float r = 1.f / (1.f + __expf(-(s7w + s6u)));   // NOTE: reference reuses eq6_u
    zu[t] = z; rg[t] = r * g[t];
    __syncthreads();
    float s8 = b8[t], s8u = bu8[t];
    for (int k = 0; k < 2 * HH; ++k) s8 += w8[(size_t)t * 2 * HH + k] * ev[k];
    for (int k = 0; k < HH; ++k) s8u += u8[(size_t)t * HH + k] * rg[k];
    float h = tanhf(s8 + s8u);
    float gn = (1.f - zu[t]) * g[t] + zu[t] * h;
    gf[t] = gn; gfb[t] = (bf16_t)gn;
}

// ----------------------------- WMMA GEMM core ------------------------------
// out = epilogue( A[M,K]bf16 @ W[N,K]^T + bias + Cin )
//   act: 0 none, 1 sigmoid, 2 tanh, 3 relu
//   if gate: out = (1-gate)*aux + gate*act(acc)      (GRU combine)
//   else if aux: out = act(acc) * aux                (r * state)
// One wave computes a 32x32 tile: 4 accumulators, A/B fragments reused 2x,
// 4 v_wmma_f32_16x16x32_bf16 per K-step. 8 waves/block cover 128(M) x 64(N).
// Ragged N (classifier heads) handled by clamping the weight-row index:
// out-of-range lanes compute garbage into columns the store guard discards,
// keeping the inner loop free of exec-mask branches.
// K multiple of 32; M multiple of 128.
__launch_bounds__(256)
__global__ void k_gemm_bf16(const bf16_t* __restrict__ A, int lda,
                            const bf16_t* __restrict__ W, int Nn, int K,
                            const float* __restrict__ bias,
                            const float* __restrict__ Cin,
                            const float* __restrict__ gate,
                            const float* __restrict__ aux,
                            float* __restrict__ out32,
                            bf16_t* __restrict__ outbf,
                            int ldc, int act) {
    const int wave = threadIdx.x >> 5;
    const int lane = threadIdx.x & 31;
    const int half = lane >> 4;
    const int ln   = lane & 15;
    const int m0 = (blockIdx.y * 4 + (wave & 3)) * 32;
    const int n0 = (blockIdx.x * 2 + (wave >> 2)) * 32;

    // A fragments: lane%16 = M row; half 0 -> K 0..15, half 1 -> K 16..31
    const bf16_t* arow0 = A + (size_t)(m0 + ln) * lda + half * 16;
    const bf16_t* arow1 = arow0 + (size_t)16 * lda;
    // B fragments: lane%16 = output column (weight row); half 0 -> K 0..15,
    // half 1 -> K 16..31. Clamp row index for ragged N (store guard covers it).
    int wr0 = n0 + ln;        if (wr0 >= Nn) wr0 = Nn - 1;
    int wr1 = n0 + 16 + ln;   if (wr1 >= Nn) wr1 = Nn - 1;
    const bf16_t* wrow0 = W + (size_t)wr0 * K + half * 16;
    const bf16_t* wrow1 = W + (size_t)wr1 * K + half * 16;

    v8f acc00 = {}, acc01 = {}, acc10 = {}, acc11 = {};

    for (int k0 = 0; k0 < K; k0 += 32) {
        v8bf a0lo = *(const v8bf*)(arow0 + k0);
        v8bf a0hi = *(const v8bf*)(arow0 + k0 + 8);
        v8bf a1lo = *(const v8bf*)(arow1 + k0);
        v8bf a1hi = *(const v8bf*)(arow1 + k0 + 8);
        v16bf a0 = __builtin_shufflevector(a0lo, a0hi, 0, 1, 2, 3, 4, 5, 6, 7,
                                                       8, 9, 10, 11, 12, 13, 14, 15);
        v16bf a1 = __builtin_shufflevector(a1lo, a1hi, 0, 1, 2, 3, 4, 5, 6, 7,
                                                       8, 9, 10, 11, 12, 13, 14, 15);
        v16bf b0 = *(const v16bf*)(wrow0 + k0);
        v16bf b1 = *(const v16bf*)(wrow1 + k0);
        acc00 = __builtin_amdgcn_wmma_f32_16x16x32_bf16(false, a0, false, b0, (short)0, acc00, false, false);
        acc01 = __builtin_amdgcn_wmma_f32_16x16x32_bf16(false, a0, false, b1, (short)0, acc01, false, false);
        acc10 = __builtin_amdgcn_wmma_f32_16x16x32_bf16(false, a1, false, b0, (short)0, acc10, false, false);
        acc11 = __builtin_amdgcn_wmma_f32_16x16x32_bf16(false, a1, false, b1, (short)0, acc11, false, false);
    }

    // epilogue: C/D layout -> element i of acc = row (tile_m + half*8 + i),
    // column (tile_n + lane%16)
    v8f* accs[4] = { &acc00, &acc01, &acc10, &acc11 };
    for (int tm = 0; tm < 2; ++tm) {
        for (int tn = 0; tn < 2; ++tn) {
            v8f& acc = *accs[tm * 2 + tn];
            const int col = n0 + tn * 16 + ln;
            if (col >= Nn) continue;
            const float bv = bias ? bias[col] : 0.0f;
            const int rbase = m0 + tm * 16 + half * 8;
            for (int i = 0; i < 8; ++i) {
                size_t idx = (size_t)(rbase + i) * ldc + col;
                float v = acc[i] + bv;
                if (Cin) v += Cin[idx];
                if (act == 1)      v = 1.f / (1.f + __expf(-v));
                else if (act == 2) v = tanhf(v);
                else if (act == 3) v = fmaxf(v, 0.f);
                if (gate) { float gz = gate[idx]; v = (1.f - gz) * aux[idx] + gz * v; }
                else if (aux) v *= aux[idx];
                if (out32) out32[idx] = v;
                if (outbf) outbf[idx] = (bf16_t)v;
            }
        }
    }
}

// ------------------------------ host driver --------------------------------

static const int WOUT[28] = {512,512,512,512,512,512,512,512,512,512,512,512,512,512,
                             512,512,512,512,512,512,512,512,512,512,512,512,OBJC,RELC};
static const int WIN[28]  = {512,512,2048,512,2048,512,2048,512,1024,512,1024,512,1024,512,
                             512,1024,1536,512,1536,512,1536,512,1024,1024,1024,1536,512,512};

extern "C" void kernel_launch(void* const* d_in, const int* in_sizes, int n_in,
                              void* d_out, int out_size, void* d_ws, size_t ws_size,
                              hipStream_t stream) {
    (void)in_sizes; (void)n_in; (void)out_size; (void)ws_size;
    const float* input = (const float*)d_in[0];
    const int*   rel   = (const int*)d_in[1];
    const int*   onb   = (const int*)d_in[2];
    auto Wf = [&](int i) { return (const float*)d_in[3 + 2 * i]; };
    auto Bf = [&](int i) { return (const float*)d_in[4 + 2 * i]; };

    // ---- scratch arena ----
    char* ws = (char*)d_ws;
    size_t off = 0;
    auto alloc = [&](size_t bytes) { void* p = ws + off; off += (bytes + 255) & ~(size_t)255; return p; };

    bf16_t* wbf[28];
    for (int i = 0; i < 28; ++i) wbf[i] = (bf16_t*)alloc((size_t)WOUT[i] * WIN[i] * 2);

    float*  edge_f = (float*) alloc((size_t)NREL * HH * 4);
    bf16_t* edge_b = (bf16_t*)alloc((size_t)NREL * HH * 2);
    bf16_t* cat_st = (bf16_t*)alloc((size_t)NREL * 2 * HH * 2);
    bf16_t* ae     = (bf16_t*)alloc((size_t)NREL * 3 * HH * 2);   // reused as [hs|ht|edge] at end
    float*  C32    = (float*) alloc((size_t)NREL * HH * 4);
    float*  ze_f   = (float*) alloc((size_t)NREL * HH * 4);
    bf16_t* reh_b  = (bf16_t*)alloc((size_t)NREL * HH * 2);       // r*edge, later output_rel
    float*  hid_f  = (float*) alloc((size_t)NOBJ * HH * 4);
    bf16_t* hid_b  = (bf16_t*)alloc((size_t)NOBJ * HH * 2);
    bf16_t* inp_b  = (bf16_t*)alloc((size_t)NOBJ * HH * 2);
    bf16_t* cat_hi = (bf16_t*)alloc((size_t)NOBJ * 2 * HH * 2);
    float*  aggS   = (float*) alloc((size_t)NOBJ * HH * 4);
    float*  aggT   = (float*) alloc((size_t)NOBJ * HH * 4);
    float*  cntS   = (float*) alloc((size_t)NOBJ * 4);
    float*  cntT   = (float*) alloc((size_t)NOBJ * 4);
    bf16_t* aggSb  = (bf16_t*)alloc((size_t)NOBJ * HH * 2);
    bf16_t* aggTb  = (bf16_t*)alloc((size_t)NOBJ * HH * 2);
    bf16_t* av     = (bf16_t*)alloc((size_t)NOBJ * 4 * HH * 2);
    float*  u3h    = (float*) alloc((size_t)NOBJ * HH * 4);
    float*  zv_f   = (float*) alloc((size_t)NOBJ * HH * 4);
    bf16_t* rh_b   = (bf16_t*)alloc((size_t)NOBJ * HH * 2);
    bf16_t* oo_b   = (bf16_t*)alloc((size_t)NOBJ * HH * 2);
    float*  emean  = (float*) alloc(HH * 4);
    float*  hmean  = (float*) alloc(HH * 4);
    bf16_t* hmeanb = (bf16_t*)alloc(HH * 2);
    float*  gf     = (float*) alloc(HH * 4);
    bf16_t* gf_b   = (bf16_t*)alloc(HH * 2);

    auto gemm = [&](const bf16_t* A, int lda, int wi, int Nn, int K,
                    const float* Cin, const float* gate, const float* aux,
                    float* o32, bf16_t* obf, int ldc, int act, int M) {
        dim3 g((Nn + 63) / 64, M / 128);
        k_gemm_bf16<<<g, 256, 0, stream>>>(A, lda, wbf[wi], Nn, K, Bf(wi),
                                           Cin, gate, aux, o32, obf, ldc, act);
    };

    // ---- one-time conversions ----
    for (int i = 0; i < 28; ++i) {
        int n = WOUT[i] * WIN[i];
        k_cvt_bf16<<<(n + 255) / 256, 256, 0, stream>>>(Wf(i), wbf[i], n);
    }
    k_init_state<<<(NOBJ * HH + 255) / 256, 256, 0, stream>>>(input, hid_f, hid_b, inp_b, NOBJ * HH);

    // ---- initial edge / global features ----
    k_gather_pair<<<NREL, 256, 0, stream>>>(hid_b, rel, onb, cat_st, 2 * HH);
    gemm(cat_st, 2 * HH, 22, HH, 2 * HH, 0, 0, 0, edge_f, edge_b, HH, 0, NREL); // edge_proj
    k_colmean<<<16, 256, 0, stream>>>(edge_f, NREL, emean, nullptr);
    k_colmean<<<16, 256, 0, stream>>>(hid_f, NOBJ, hmean, hmeanb);
    k_matvec2<<<2, 256, 0, stream>>>(Wf(23), Bf(23), emean, HH, hmean, HH, gf, gf_b);

    for (int it = 0; it < 3; ++it) {
        // ---- edge update ----
        k_gather_pair<<<NREL, 256, 0, stream>>>(hid_b, rel, onb, cat_st, 2 * HH);
        gemm(edge_b, HH, 14, HH, HH, 0, 0, 0, 0, ae, 3 * HH, 0, NREL);              // edge_trans_a
        gemm(cat_st, 2 * HH, 15, HH, 2 * HH, 0, 0, 0, 0, ae + HH, 3 * HH, 0, NREL); // edge_trans_b
        k_fill_rows<<<NREL, 256, 0, stream>>>(ae + 2 * HH, 3 * HH, gf_b);           // global bcast
        gemm(ae, 3 * HH, 16, HH, 3 * HH, 0, 0, 0, C32, 0, HH, 0, NREL);             // W9@ae
        gemm(edge_b, HH, 17, HH, HH, C32, 0, 0, ze_f, 0, HH, 1, NREL);              // +U9@e, sigmoid
        gemm(ae, 3 * HH, 18, HH, 3 * HH, 0, 0, 0, C32, 0, HH, 0, NREL);             // W10@ae
        gemm(edge_b, HH, 19, HH, HH, C32, 0, edge_f, 0, reh_b, HH, 1, NREL);        // re*edge -> bf16
        gemm(ae, 3 * HH, 20, HH, 3 * HH, 0, 0, 0, C32, 0, HH, 0, NREL);             // W11@ae
        gemm(reh_b, HH, 21, HH, HH, C32, ze_f, edge_f, edge_f, edge_b, HH, 2, NREL); // tanh + GRU

        // ---- segment means ----
        k_zero_f32<<<1024, 256, 0, stream>>>(aggS, NOBJ * HH);
        k_zero_f32<<<1024, 256, 0, stream>>>(aggT, NOBJ * HH);
        k_zero_f32<<<8, 256, 0, stream>>>(cntS, NOBJ);
        k_zero_f32<<<8, 256, 0, stream>>>(cntT, NOBJ);
        k_scatter<<<NREL, 256, 0, stream>>>(edge_f, rel, onb, aggS, aggT, cntS, cntT);
        k_segnorm<<<(NOBJ * HH + 255) / 256, 256, 0, stream>>>(aggS, cntS, aggSb, NOBJ * HH);
        k_segnorm<<<(NOBJ * HH + 255) / 256, 256, 0, stream>>>(aggT, cntT, aggTb, NOBJ * HH);

        // ---- node update ----
        k_colmean<<<16, 256, 0, stream>>>(hid_f, NOBJ, hmean, hmeanb);  // pre-update mean
        gemm(aggSb, HH, 0, HH, HH, 0, 0, 0, 0, av, 4 * HH, 0, NOBJ);                // node_trans_a
        gemm(aggTb, HH, 1, HH, HH, 0, 0, 0, 0, av + HH, 4 * HH, 0, NOBJ);           // node_trans_b
        k_fill_rows<<<NOBJ, 256, 0, stream>>>(av + 2 * HH, 4 * HH, hmeanb);
        k_fill_rows<<<NOBJ, 256, 0, stream>>>(av + 3 * HH, 4 * HH, gf_b);
        gemm(hid_b, HH, 3, HH, HH, 0, 0, 0, u3h, 0, HH, 0, NOBJ);                   // U3@h (shared)
        gemm(av, 4 * HH, 2, HH, 4 * HH, u3h, 0, 0, zv_f, 0, HH, 1, NOBJ);           // zv
        gemm(av, 4 * HH, 4, HH, 4 * HH, u3h, 0, hid_f, 0, rh_b, HH, 1, NOBJ);       // rv*h (ref bug: U3)
        gemm(av, 4 * HH, 6, HH, 4 * HH, 0, 0, 0, C32, 0, HH, 0, NOBJ);              // W5@av
        gemm(rh_b, HH, 7, HH, HH, C32, zv_f, hid_f, hid_f, hid_b, HH, 2, NOBJ);     // tanh + GRU

        // ---- global update ----
        k_colmean<<<16, 256, 0, stream>>>(edge_f, NREL, emean, nullptr);
        k_colmean<<<16, 256, 0, stream>>>(hid_f, NOBJ, hmean, nullptr);
        k_global_gru<<<1, 512, 0, stream>>>(emean, hmean,
                                            Wf(8), Bf(8), Wf(9), Bf(9), Wf(10), Bf(10),
                                            Wf(12), Bf(12), Wf(13), Bf(13), gf, gf_b);
    }

    // ---- output heads ----
    k_copy_rows<<<NOBJ, 256, 0, stream>>>(cat_hi, 2 * HH, hid_b, HH);
    k_copy_rows<<<NOBJ, 256, 0, stream>>>(cat_hi + HH, 2 * HH, inp_b, HH);
    gemm(cat_hi, 2 * HH, 24, HH, 2 * HH, 0, 0, 0, 0, oo_b, HH, 3, NOBJ);            // relu(output_obj)
    gemm(oo_b, HH, 26, OBJC, HH, 0, 0, 0, (float*)d_out, 0, OBJC, 0, NOBJ);         // obj_dists

    k_gather_pair<<<NREL, 256, 0, stream>>>(hid_b, rel, onb, ae, 3 * HH);
    k_copy_rows<<<NREL, 256, 0, stream>>>(ae + 2 * HH, 3 * HH, edge_b, HH);
    gemm(ae, 3 * HH, 25, HH, 3 * HH, 0, 0, 0, 0, reh_b, HH, 0, NREL);               // output_rel
    gemm(reh_b, HH, 27, RELC, HH, 0, 0, 0,
         (float*)d_out + (size_t)NOBJ * OBJC, 0, RELC, 0, NREL);                    // rel_dists
}
